// MultiHeadAttention_33354716021190
// MI455X (gfx1250) — matrix operand order repaired and checked by reading, not compiled, not run on hardware
//
#include <hip/hip_runtime.h>

typedef __attribute__((ext_vector_type(16))) _Float16 v16h;
typedef __attribute__((ext_vector_type(8)))  _Float16 v8h;
typedef __attribute__((ext_vector_type(8)))  float    v8f;
typedef __attribute__((ext_vector_type(4)))  float    v4f;

constexpr int B_ = 4, S_ = 2048, NH_ = 16, HD_ = 64, HID_ = 1024;

__device__ __forceinline__ v8f wmma16(v16h a, v16h b, v8f c) {
  return __builtin_amdgcn_wmma_f32_16x16x32_f16(false, a, false, b, (short)0, c,
                                                false, false);
}

// A-matrix fragment (16x32 f16): lane l -> row l&15; K chunk {0..7,16..23} for
// lanes 0-15, {8..15,24..31} for lanes 16-31.  base points at [row0][k0].
__device__ __forceinline__ v16h fragA(const _Float16* base, int stride, int lane) {
  int r = lane & 15;
  int c = (lane & 16) ? 8 : 0;
  v8h lo = *(const v8h*)(base + r * stride + c);
  v8h hi = *(const v8h*)(base + r * stride + c + 16);
  return __builtin_shufflevector(lo, hi, 0,1,2,3,4,5,6,7,8,9,10,11,12,13,14,15);
}

// B-matrix fragment (32x16 f16): lane n -> col n&15; K {0..7,16..23} (lanes 0-15)
// or {8..15,24..31} (lanes 16-31).  LDS tile stored as [col][K] (K contiguous).
__device__ __forceinline__ v16h fragB(const _Float16* base, int stride, int lane) {
  int r = lane & 15;
  int c = (lane & 16) ? 8 : 0;
  v8h lo = *(const v8h*)(base + r * stride + c);
  v8h hi = *(const v8h*)(base + r * stride + c + 16);
  return __builtin_shufflevector(lo, hi, 0,1,2,3,4,5,6,7,8,9,10,11,12,13,14,15);
}

// Async global->LDS copy of 32 bytes per lane (two B128 moves; INST_OFFSET is
// applied to both the LDS and global addresses per the CDNA5 ISA). Tracked by
// ASYNCcnt; pair with async_wait0() before the workgroup barrier.
__device__ __forceinline__ void async_copy32B(unsigned lds_off, const void* gptr) {
  asm volatile(
      "global_load_async_to_lds_b128 %0, %1, off\n\t"
      "global_load_async_to_lds_b128 %0, %1, off offset:16"
      :
      : "v"(lds_off), "v"(gptr)
      : "memory");
}
__device__ __forceinline__ void async_wait0() {
  asm volatile("s_wait_asynccnt 0x0" ::: "memory");
}
__device__ __forceinline__ unsigned lds_addr(const void* p) {
  return (unsigned)(size_t)p;  // flat LDS pointer carries LDS offset in [31:0]
}

// ---------------------------------------------------------------------------
// QKV projection: Y = (X @ W^T + b) * outScale, written in f16.
// vmode 0: Y layout (B, NH, S, HD)   (for Q and K)
// vmode 1: Y layout (B, NH, HD, S)   (V transposed for the PV WMMA B-operand)
// Tile: 128x128, K-step 32. 256 threads = 8 waves in a 2x4 grid (64x32 each).
// fp32 inputs are converted to f16 in flight (so no async path here).
// ---------------------------------------------------------------------------
__global__ __launch_bounds__(256) void mha_proj_kernel(
    const float* __restrict__ X, const float* __restrict__ W,
    const float* __restrict__ bias, _Float16* __restrict__ Y,
    int vmode, float outScale) {
  __shared__ _Float16 sA[128 * 40];
  __shared__ _Float16 sB[128 * 40];
  const int t = threadIdx.x, lane = t & 31, wid = t >> 5;
  const int m0 = blockIdx.x * 128, n0 = blockIdx.y * 128;
  const int wr = (wid >> 2) * 64, wc = (wid & 3) * 32;
  const int lrow = t >> 1, lcol = (t & 1) * 16;

  v8f acc[4][2] = {};

  for (int k0 = 0; k0 < HID_; k0 += 32) {
    {
      const float* sx = X + (size_t)(m0 + lrow) * HID_ + k0 + lcol;
      v4f a0 = *(const v4f*)(sx), a1 = *(const v4f*)(sx + 4);
      v4f a2 = *(const v4f*)(sx + 8), a3 = *(const v4f*)(sx + 12);
      v8h h0, h1;
#pragma unroll
      for (int i = 0; i < 4; i++) {
        h0[i] = (_Float16)a0[i]; h0[4 + i] = (_Float16)a1[i];
        h1[i] = (_Float16)a2[i]; h1[4 + i] = (_Float16)a3[i];
      }
      *(v8h*)(sA + lrow * 40 + lcol) = h0;
      *(v8h*)(sA + lrow * 40 + lcol + 8) = h1;

      const float* sw = W + (size_t)(n0 + lrow) * HID_ + k0 + lcol;
      v4f b0 = *(const v4f*)(sw), b1 = *(const v4f*)(sw + 4);
      v4f b2 = *(const v4f*)(sw + 8), b3 = *(const v4f*)(sw + 12);
      v8h g0, g1;
#pragma unroll
      for (int i = 0; i < 4; i++) {
        g0[i] = (_Float16)b0[i]; g0[4 + i] = (_Float16)b1[i];
        g1[i] = (_Float16)b2[i]; g1[4 + i] = (_Float16)b3[i];
      }
      *(v8h*)(sB + lrow * 40 + lcol) = g0;
      *(v8h*)(sB + lrow * 40 + lcol + 8) = g1;
    }
    __syncthreads();

    v16h af[4], bf[2];
#pragma unroll
    for (int tr = 0; tr < 4; tr++) af[tr] = fragA(sA + (wr + tr * 16) * 40, 40, lane);
#pragma unroll
    for (int tc = 0; tc < 2; tc++) bf[tc] = fragB(sB + (wc + tc * 16) * 40, 40, lane);
#pragma unroll
    for (int tr = 0; tr < 4; tr++)
#pragma unroll
      for (int tc = 0; tc < 2; tc++) acc[tr][tc] = wmma16(af[tr], bf[tc], acc[tr][tc]);
    __syncthreads();
  }

  const int hi = lane >> 4, lo = lane & 15;
#pragma unroll
  for (int tc = 0; tc < 2; tc++) {
    const int n = n0 + wc + tc * 16 + lo;
    const float bv = bias[n];
    const int h = n >> 6, d = n & 63;
#pragma unroll
    for (int tr = 0; tr < 4; tr++) {
#pragma unroll
      for (int i = 0; i < 8; i++) {
        const int m = m0 + wr + tr * 16 + i + 8 * hi;
        const int bb = m >> 11, s = m & (S_ - 1);
        const float val = (acc[tr][tc][i] + bv) * outScale;
        size_t idx = vmode
            ? ((size_t)((bb * NH_ + h) * HD_ + d)) * S_ + s
            : ((size_t)((bb * NH_ + h) * S_ + s)) * HD_ + d;
        Y[idx] = (_Float16)val;
      }
    }
  }
}

// ---------------------------------------------------------------------------
// Flash attention: one workgroup per (b, h, 128-query block); each of the 8
// waves owns 16 query rows. Keys staged in LDS blocks of 64 via async
// global->LDS copies (ASYNCcnt). Q pre-scaled by 1/sqrt(HD).
// ---------------------------------------------------------------------------
__global__ __launch_bounds__(256) void mha_attn_kernel(
    const _Float16* __restrict__ Qs, const _Float16* __restrict__ Ks,
    const _Float16* __restrict__ Vt, const unsigned char* __restrict__ mask,
    _Float16* __restrict__ Ctx) {
  __shared__ _Float16 sK[64 * 72];        // [key][d]
  __shared__ _Float16 sV[64 * 72];        // [d][key]
  __shared__ _Float16 sP[8 * 16 * 72];    // per-wave P bounce [row][key]
  const int t = threadIdx.x, lane = t & 31, wid = t >> 5;
  const int qb = blockIdx.x, h = blockIdx.y, b = blockIdx.z;
  const size_t bh = (size_t)(b * NH_ + h);
  const _Float16* Qbase = Qs + bh * S_ * HD_;
  const _Float16* Kbase = Ks + bh * S_ * HD_;
  const _Float16* Vbase = Vt + bh * HD_ * S_;
  const int q0 = qb * 128 + wid * 16;
  const int hi = lane >> 4, lo = lane & 15;

  v16h qf[2];
#pragma unroll
  for (int f = 0; f < 2; f++) {
    const int c = f * 32 + ((lane & 16) ? 8 : 0);
    const _Float16* p = Qbase + (size_t)(q0 + (lane & 15)) * HD_ + c;
    v8h l0 = *(const v8h*)p;
    v8h h0 = *(const v8h*)(p + 16);
    qf[f] = __builtin_shufflevector(l0, h0, 0,1,2,3,4,5,6,7,8,9,10,11,12,13,14,15);
  }

  float m_[8], l_[8];
  v8f oacc[4] = {};
#pragma unroll
  for (int i = 0; i < 8; i++) { m_[i] = -3e38f; l_[i] = 0.f; }

  const int lrow = t >> 2, lcol = (t & 3) * 16;
  _Float16* spw = sP + wid * (16 * 72);
  const unsigned ldsK = lds_addr(sK + lrow * 72 + lcol);
  const unsigned ldsV = lds_addr(sV + lrow * 72 + lcol);

  for (int kb = 0; kb < S_ / 64; kb++) {
    __syncthreads();
    // Stage K block (64 keys x 64 d) and V block (64 d x 64 keys) async.
    async_copy32B(ldsK, Kbase + (size_t)(kb * 64 + lrow) * HD_ + lcol);
    async_copy32B(ldsV, Vbase + (size_t)lrow * S_ + kb * 64 + lcol);
    async_wait0();
    __syncthreads();

    // Scores: 16 q-rows x 64 keys (already scaled via Q).
    v8f sacc[4] = {};
#pragma unroll
    for (int tc = 0; tc < 4; tc++)
#pragma unroll
      for (int f = 0; f < 2; f++) {
        v16h bf = fragB(sK + (tc * 16) * 72 + f * 32, 72, lane);
        sacc[tc] = wmma16(qf[f], bf, sacc[tc]);
      }

    // Mask (reference: where(mask, -1e9, s)).
#pragma unroll
    for (int tc = 0; tc < 4; tc++)
#pragma unroll
      for (int i = 0; i < 8; i++) {
        const int qr = q0 + i + 8 * hi;
        const int key = kb * 64 + tc * 16 + lo;
        if (mask[((size_t)b * S_ + qr) * S_ + key]) sacc[tc][i] = -1.0e9f;
      }

    // Online softmax update.
    float fac[8];
#pragma unroll
    for (int i = 0; i < 8; i++) {
      float mx = fmaxf(fmaxf(sacc[0][i], sacc[1][i]),
                       fmaxf(sacc[2][i], sacc[3][i]));
#pragma unroll
      for (int off = 1; off < 16; off <<= 1) mx = fmaxf(mx, __shfl_xor(mx, off, 32));
      const float mn = fmaxf(m_[i], mx);
      fac[i] = __expf(m_[i] - mn);
      m_[i] = mn;
      float rs = 0.f;
#pragma unroll
      for (int tc = 0; tc < 4; tc++) {
        const float p = __expf(sacc[tc][i] - mn);
        sacc[tc][i] = p;
        rs += p;
      }
#pragma unroll
      for (int off = 1; off < 16; off <<= 1) rs += __shfl_xor(rs, off, 32);
      l_[i] = l_[i] * fac[i] + rs;
    }
#pragma unroll
    for (int td = 0; td < 4; td++)
#pragma unroll
      for (int i = 0; i < 8; i++) oacc[td][i] *= fac[i];

    // Bounce P through LDS to re-layout C-layout -> A-layout.
#pragma unroll
    for (int tc = 0; tc < 4; tc++)
#pragma unroll
      for (int i = 0; i < 8; i++)
        spw[(i + 8 * hi) * 72 + tc * 16 + lo] = (_Float16)sacc[tc][i];
    __syncthreads();

    v16h pf[2];
#pragma unroll
    for (int f = 0; f < 2; f++) pf[f] = fragA(spw + f * 32, 72, lane);
#pragma unroll
    for (int td = 0; td < 4; td++)
#pragma unroll
      for (int f = 0; f < 2; f++) {
        v16h bf = fragB(sV + (td * 16) * 72 + f * 32, 72, lane);
        oacc[td] = wmma16(pf[f], bf, oacc[td]);
      }
  }

  // Normalize and store context (B, S, NH*HD) f16.
#pragma unroll
  for (int td = 0; td < 4; td++)
#pragma unroll
    for (int i = 0; i < 8; i++) {
      const int row = q0 + i + 8 * hi;
      const int col = h * HD_ + td * 16 + lo;
      Ctx[((size_t)b * S_ + row) * (NH_ * HD_) + col] =
          (_Float16)(oacc[td][i] / l_[i]);
    }
}

// ---------------------------------------------------------------------------
// Head-0 attention probabilities (top_attn), two-pass: pass 1 gets row max and
// sum, pass 2 recomputes scores and writes exp(s-m)/l as fp32.
// ---------------------------------------------------------------------------
__global__ __launch_bounds__(256) void mha_topattn_kernel(
    const _Float16* __restrict__ Qs, const _Float16* __restrict__ Ks,
    const unsigned char* __restrict__ mask, float* __restrict__ Out) {
  __shared__ _Float16 sK[64 * 72];
  const int t = threadIdx.x, lane = t & 31, wid = t >> 5;
  const int qb = blockIdx.x, b = blockIdx.y;
  const size_t bh = (size_t)(b * NH_);  // head 0
  const _Float16* Qbase = Qs + bh * S_ * HD_;
  const _Float16* Kbase = Ks + bh * S_ * HD_;
  const int q0 = qb * 128 + wid * 16;
  const int hi = lane >> 4, lo = lane & 15;

  v16h qf[2];
#pragma unroll
  for (int f = 0; f < 2; f++) {
    const int c = f * 32 + ((lane & 16) ? 8 : 0);
    const _Float16* p = Qbase + (size_t)(q0 + (lane & 15)) * HD_ + c;
    v8h l0 = *(const v8h*)p;
    v8h h0 = *(const v8h*)(p + 16);
    qf[f] = __builtin_shufflevector(l0, h0, 0,1,2,3,4,5,6,7,8,9,10,11,12,13,14,15);
  }

  float m_[8], l_[8];
#pragma unroll
  for (int i = 0; i < 8; i++) { m_[i] = -3e38f; l_[i] = 0.f; }
  const int lrow = t >> 2, lcol = (t & 3) * 16;
  const unsigned ldsK = lds_addr(sK + lrow * 72 + lcol);

  for (int pass = 0; pass < 2; pass++) {
    for (int kb = 0; kb < S_ / 64; kb++) {
      __syncthreads();
      async_copy32B(ldsK, Kbase + (size_t)(kb * 64 + lrow) * HD_ + lcol);
      async_wait0();
      __syncthreads();

      v8f sacc[4] = {};
#pragma unroll
      for (int tc = 0; tc < 4; tc++)
#pragma unroll
        for (int f = 0; f < 2; f++) {
          v16h bf = fragB(sK + (tc * 16) * 72 + f * 32, 72, lane);
          sacc[tc] = wmma16(qf[f], bf, sacc[tc]);
        }
#pragma unroll
      for (int tc = 0; tc < 4; tc++)
#pragma unroll
        for (int i = 0; i < 8; i++) {
          const int qr = q0 + i + 8 * hi;
          const int key = kb * 64 + tc * 16 + lo;
          if (mask[((size_t)b * S_ + qr) * S_ + key]) sacc[tc][i] = -1.0e9f;
        }

      if (pass == 0) {
#pragma unroll
        for (int i = 0; i < 8; i++) {
          float mx = fmaxf(fmaxf(sacc[0][i], sacc[1][i]),
                           fmaxf(sacc[2][i], sacc[3][i]));
#pragma unroll
          for (int off = 1; off < 16; off <<= 1)
            mx = fmaxf(mx, __shfl_xor(mx, off, 32));
          const float mn = fmaxf(m_[i], mx);
          const float fac = __expf(m_[i] - mn);
          m_[i] = mn;
          float rs = 0.f;
#pragma unroll
          for (int tc = 0; tc < 4; tc++) rs += __expf(sacc[tc][i] - mn);
#pragma unroll
          for (int off = 1; off < 16; off <<= 1) rs += __shfl_xor(rs, off, 32);
          l_[i] = l_[i] * fac + rs;
        }
      } else {
#pragma unroll
        for (int tc = 0; tc < 4; tc++)
#pragma unroll
          for (int i = 0; i < 8; i++) {
            const int qr = q0 + i + 8 * hi;
            const int key = kb * 64 + tc * 16 + lo;
            Out[((size_t)b * S_ + qr) * S_ + key] =
                __expf(sacc[tc][i] - m_[i]) / l_[i];
          }
      }
    }
  }
}

// ---------------------------------------------------------------------------
// Output projection: Out = Ctx(f16) @ W_o^T + b_o, fp32 out, plain layout.
// Ctx tile is staged with async global->LDS copies; W_o needs fp32->f16
// conversion so it keeps the VALU path.
// ---------------------------------------------------------------------------
__global__ __launch_bounds__(256) void mha_outproj_kernel(
    const _Float16* __restrict__ Ctx, const float* __restrict__ W,
    const float* __restrict__ bias, float* __restrict__ Out) {
  __shared__ _Float16 sA[128 * 40];
  __shared__ _Float16 sB[128 * 40];
  const int t = threadIdx.x, lane = t & 31, wid = t >> 5;
  const int m0 = blockIdx.x * 128, n0 = blockIdx.y * 128;
  const int wr = (wid >> 2) * 64, wc = (wid & 3) * 32;
  const int lrow = t >> 1, lcol = (t & 1) * 16;
  const unsigned ldsA = lds_addr(sA + lrow * 40 + lcol);

  v8f acc[4][2] = {};

  for (int k0 = 0; k0 < HID_; k0 += 32) {
    async_copy32B(ldsA, Ctx + (size_t)(m0 + lrow) * HID_ + k0 + lcol);
    {
      const float* sw = W + (size_t)(n0 + lrow) * HID_ + k0 + lcol;
      v4f b0 = *(const v4f*)(sw), b1 = *(const v4f*)(sw + 4);
      v4f b2 = *(const v4f*)(sw + 8), b3 = *(const v4f*)(sw + 12);
      v8h g0, g1;
#pragma unroll
      for (int i = 0; i < 4; i++) {
        g0[i] = (_Float16)b0[i]; g0[4 + i] = (_Float16)b1[i];
        g1[i] = (_Float16)b2[i]; g1[4 + i] = (_Float16)b3[i];
      }
      *(v8h*)(sB + lrow * 40 + lcol) = g0;
      *(v8h*)(sB + lrow * 40 + lcol + 8) = g1;
    }
    async_wait0();
    __syncthreads();

    v16h af[4], bf[2];
#pragma unroll
    for (int tr = 0; tr < 4; tr++) af[tr] = fragA(sA + (wr + tr * 16) * 40, 40, lane);
#pragma unroll
    for (int tc = 0; tc < 2; tc++) bf[tc] = fragB(sB + (wc + tc * 16) * 40, 40, lane);
#pragma unroll
    for (int tr = 0; tr < 4; tr++)
#pragma unroll
      for (int tc = 0; tc < 2; tc++) acc[tr][tc] = wmma16(af[tr], bf[tc], acc[tr][tc]);
    __syncthreads();
  }

  const int hi = lane >> 4, lo = lane & 15;
#pragma unroll
  for (int tc = 0; tc < 2; tc++) {
    const int n = n0 + wc + tc * 16 + lo;
    const float bv = bias[n];
#pragma unroll
    for (int tr = 0; tr < 4; tr++)
#pragma unroll
      for (int i = 0; i < 8; i++) {
        const int m = m0 + wr + tr * 16 + i + 8 * hi;
        Out[(size_t)m * HID_ + n] = acc[tr][tc][i] + bv;
      }
  }
}

extern "C" void kernel_launch(void* const* d_in, const int* in_sizes, int n_in,
                              void* d_out, int out_size, void* d_ws, size_t ws_size,
                              hipStream_t stream) {
  (void)in_sizes; (void)n_in; (void)out_size; (void)ws_size;
  const float* q  = (const float*)d_in[0];
  const float* k  = (const float*)d_in[1];
  const float* v  = (const float*)d_in[2];
  const unsigned char* mask = (const unsigned char*)d_in[3];
  const float* Wq = (const float*)d_in[4];
  const float* bq = (const float*)d_in[5];
  const float* Wk = (const float*)d_in[6];
  const float* bk = (const float*)d_in[7];
  const float* Wv = (const float*)d_in[8];
  const float* bv = (const float*)d_in[9];
  const float* Wo = (const float*)d_in[10];
  const float* bo = (const float*)d_in[11];

  float* out = (float*)d_out;
  float* top = out + (size_t)B_ * S_ * HID_;  // attn[:,0,:,:]

  const size_t HALF_TENSOR = (size_t)B_ * NH_ * S_ * HD_;  // 8,388,608 halfs
  _Float16* qs  = (_Float16*)d_ws;
  _Float16* ks  = qs + HALF_TENSOR;
  _Float16* vt  = ks + HALF_TENSOR;
  _Float16* ctx = vt + HALF_TENSOR;

  dim3 blk(256, 1, 1);
  dim3 gProj(B_ * S_ / 128, HID_ / 128, 1);  // 64 x 8

  // Q projection carries the 1/sqrt(HD) = 0.125 scale (exact in f16).
  mha_proj_kernel<<<gProj, blk, 0, stream>>>(q, Wq, bq, qs, 0, 0.125f);
  mha_proj_kernel<<<gProj, blk, 0, stream>>>(k, Wk, bk, ks, 0, 1.0f);
  mha_proj_kernel<<<gProj, blk, 0, stream>>>(v, Wv, bv, vt, 1, 1.0f);

  mha_attn_kernel<<<dim3(S_ / 128, NH_, B_), blk, 0, stream>>>(qs, ks, vt, mask, ctx);
  mha_topattn_kernel<<<dim3(S_ / 128, B_, 1), blk, 0, stream>>>(qs, ks, mask, top);
  mha_outproj_kernel<<<gProj, blk, 0, stream>>>(ctx, Wo, bo, out);
}
